// BiLSTM_90005334655619
// MI455X (gfx1250) — matrix operand order repaired and checked by reading, not compiled, not run on hardware
//
#include <hip/hip_runtime.h>
#include <hip/hip_bf16.h>
#include <stdint.h>
#include <stddef.h>

// ---------------------------------------------------------------------------
// BiLSTM for MI455X (gfx1250): bf16 WMMA with f32 accumulation.
//   Phase 1: convert x / weights to bf16, fold biases (bih+bhh).
//   Phase 2: big parallel WMMA GEMM: Gx[t,b,:] = x_t @ Wih^T + bias (both dirs;
//            backward dir indexes x flipped in (time, batch) == flat 65535-m).
//   Phase 3: persistent scan kernel: one 32-wave workgroup per direction.
//            - Whh fragments resident in VGPRs for the whole 2048-step loop
//            - h (bf16 A-operand) + gate staging in LDS
//            - h/c carry state in per-thread registers (exact f32)
//            - next step's Gx slice streamed into an LDS double buffer with
//              GLOBAL_LOAD_ASYNC_TO_LDS_B128 (ASYNCcnt), hidden under the WMMAs
// ---------------------------------------------------------------------------

typedef __attribute__((ext_vector_type(16))) __bf16 v16bf;
typedef __attribute__((ext_vector_type(8)))  __bf16 v8bf;
typedef __attribute__((ext_vector_type(8)))  float  v8f;

#define T_STEPS 2048
#define NB      32            // batch
#define NI      256           // input size
#define NH      256           // hidden size
#define NG      1024          // 4*H (gates)
#define MROWS   (T_STEPS * NB)  // 65536 rows of the input GEMM

// ---------------------------------------------------------------------------
// WMMA fragment helpers (16x16x32 bf16). See round-1 notes for lane layouts.
// ---------------------------------------------------------------------------
static __device__ __forceinline__ v16bf wmma_a_frag(const __bf16* __restrict__ p) {
  v8bf lo = *(const v8bf*)p;
  v8bf hi = *(const v8bf*)(p + 16);
  v16bf r;
#pragma unroll
  for (int i = 0; i < 8; ++i) { r[i] = lo[i]; r[i + 8] = hi[i]; }
  return r;
}

static __device__ __forceinline__ v16bf wmma_b_frag(const __bf16* __restrict__ W,
                                                    int n0, int k0, int lane, int ldk) {
  const int n  = n0 + (lane & 15);
  const int kh = (lane >> 4) << 3;
  const __bf16* p = W + (size_t)n * ldk + k0 + kh;
  v8bf lo = *(const v8bf*)p;
  v8bf hi = *(const v8bf*)(p + 16);
  v16bf r;
#pragma unroll
  for (int i = 0; i < 8; ++i) { r[i] = lo[i]; r[i + 8] = hi[i]; }
  return r;
}

static __device__ __forceinline__ v8f wmma_bf16(v16bf a, v16bf b, v8f c) {
  return __builtin_amdgcn_wmma_f32_16x16x32_bf16(false, a, false, b, (short)0, c,
                                                 false, false);
}

static __device__ __forceinline__ float sigmoid_(float x) {
  return 1.0f / (1.0f + __expf(-x));
}
static __device__ __forceinline__ float tanh_(float x) {
  return 1.0f - 2.0f / (1.0f + __expf(2.0f * x));
}

// ---------------------------------------------------------------------------
// Async global -> LDS copy (16 B per lane) + ASYNCcnt wait.
// Inline asm (portable across toolchains; bypasses clang builtin pointer
// address-space typing). LDS operand = low 32 bits of the generic pointer
// (ISA 10.2: generic LDS addr[31:0] is the LDS address).
// ---------------------------------------------------------------------------
static __device__ __forceinline__ void async_copy_b128(void* lds, const void* g) {
  const uint32_t loff = (uint32_t)(uintptr_t)lds;
  asm volatile("global_load_async_to_lds_b128 %0, %1, off"
               :: "v"(loff), "v"(g) : "memory");
}

static __device__ __forceinline__ void wait_asynccnt_4() {
#if __has_builtin(__builtin_amdgcn_s_wait_asynccnt)
  __builtin_amdgcn_s_wait_asynccnt(4);
#else
  asm volatile("s_wait_asynccnt 4" ::: "memory");
#endif
}

// ---------------------------------------------------------------------------
// Phase 1 kernels
// ---------------------------------------------------------------------------
__global__ void lstm_cvt_bf16(const float* __restrict__ in, __bf16* __restrict__ out,
                              int n) {
  int i = blockIdx.x * blockDim.x + threadIdx.x;
  const int stride = gridDim.x * blockDim.x;
  for (; i < n; i += stride) out[i] = (__bf16)in[i];
}

__global__ void lstm_bias_sum(const float* __restrict__ a, const float* __restrict__ b,
                              float* __restrict__ o, int n) {
  int i = blockIdx.x * blockDim.x + threadIdx.x;
  if (i < n) o[i] = a[i] + b[i];
}

// ---------------------------------------------------------------------------
// Phase 2: Gx[dir][m][n] = x_row(dir,m) @ Wih[dir]^T + (bih+bhh)[dir][n]
// grid = (MROWS/64, NG/64, 2); block = 256 (8 waves); each wave: 16x32 tile.
// ---------------------------------------------------------------------------
__global__ __launch_bounds__(256)
void lstm_gemm_x(const __bf16* __restrict__ xb,
                 const __bf16* __restrict__ Wih_f, const __bf16* __restrict__ Wih_b,
                 const float*  __restrict__ bsum_f, const float* __restrict__ bsum_b,
                 __bf16* __restrict__ Gx_f, __bf16* __restrict__ Gx_b) {
  const int dir = blockIdx.z;
  const __bf16* W  = dir ? Wih_b  : Wih_f;
  const float*  bs = dir ? bsum_b : bsum_f;
  __bf16*       Gx = dir ? Gx_b   : Gx_f;

  const int lane  = threadIdx.x & 31;
  const int w     = threadIdx.x >> 5;
  const int mbase = blockIdx.x * 64 + ((w & 3) << 4);
  const int nbase = blockIdx.y * 64 + ((w >> 2) << 5);

  // flip(time, batch) of a [T,B] flat index == 65535 - m
  const int    mrow = mbase + (lane & 15);
  const size_t xrow = dir ? (size_t)(MROWS - 1 - mrow) : (size_t)mrow;
  const __bf16* arow = xb + xrow * NI + ((lane >> 4) << 3);

  v8f acc0 = {}; v8f acc1 = {};
#pragma unroll
  for (int kk = 0; kk < 8; ++kk) {
    const int k0 = kk * 32;
    v16bf a  = wmma_a_frag(arow + k0);
    v16bf b0 = wmma_b_frag(W, nbase,      k0, lane, NI);
    v16bf b1 = wmma_b_frag(W, nbase + 16, k0, lane, NI);
    acc0 = wmma_bf16(a, b0, acc0);
    acc1 = wmma_bf16(a, b1, acc1);
  }

  const int   nc = nbase + (lane & 15);
  const float bsum0 = bs[nc];
  const float bsum1 = bs[nc + 16];
#pragma unroll
  for (int r = 0; r < 8; ++r) {
    const int M = mbase + r + ((lane >> 4) << 3);
    Gx[(size_t)M * NG + nc]      = (__bf16)(acc0[r] + bsum0);
    Gx[(size_t)M * NG + nc + 16] = (__bf16)(acc1[r] + bsum1);
  }
}

// ---------------------------------------------------------------------------
// Phase 3: persistent scan. grid = 2 (dir), block = 1024 (32 waves).
// ---------------------------------------------------------------------------
__global__ __launch_bounds__(1024)
void lstm_scan(const __bf16* __restrict__ Gx_f, const __bf16* __restrict__ Gx_b,
               const __bf16* __restrict__ Whh_f, const __bf16* __restrict__ Whh_b,
               const float* __restrict__ h0, const float* __restrict__ c0,
               const int* __restrict__ length, float* __restrict__ out) {
  const int dir = blockIdx.x;
  const __bf16* Gx = dir ? Gx_b  : Gx_f;
  const __bf16* W  = dir ? Whh_b : Whh_f;

  __shared__ __bf16 h_bf[NB * NH];          // 16 KB   (WMMA A operand)
  __shared__ float  gates[NB * NG];         // 128 KB  (recurrent matmul result)
  __shared__ __bf16 gxbuf[2][NB * NG];      // 128 KB  (async Gx double buffer)
  __shared__ int    len_s[NB];

  const int tid  = threadIdx.x;
  const int lane = tid & 31;
  const int w    = tid >> 5;
  const int nbase = w << 5;

  // ---- preload this wave's Whh fragments into registers (loop-invariant) ----
  v16bf wB0[8], wB1[8];
#pragma unroll
  for (int kk = 0; kk < 8; ++kk) {
    wB0[kk] = wmma_b_frag(W, nbase,      kk * 32, lane, NH);
    wB1[kk] = wmma_b_frag(W, nbase + 16, kk * 32, lane, NH);
  }

  // ---- per-thread carry state: thread owns elements idx = tid + e*1024 ----
  float hreg[8], creg[8];
  if (tid < NB) len_s[tid] = length[tid];
#pragma unroll
  for (int e = 0; e < 8; ++e) {
    const int idx = tid + e * 1024;          // NB*NH = 8192 elements
    hreg[e] = h0[idx];
    creg[e] = c0[idx];
    h_bf[idx] = (__bf16)hreg[e];
  }

  // ---- kick off async load of Gx slice for t = 0 into gxbuf[0] ----
  {
    const char* g0 = (const char*)Gx;
    char*       l0 = (char*)&gxbuf[0][0];
#pragma unroll
    for (int i = 0; i < 4; ++i)
      async_copy_b128(l0 + i * 16384 + tid * 16,
                      g0 + i * 16384 + (size_t)tid * 16);
  }
  __syncthreads();

  float* out_h = out;
  float* out_c = out + (size_t)T_STEPS * 2 * NB * NH;

  for (int t = 0; t < T_STEPS; ++t) {
    // ---- issue async loads for step t+1 into the other buffer ----
    {
      const int tn = (t + 1 < T_STEPS) ? (t + 1) : (T_STEPS - 1); // clamp: keeps
      const char* gn = (const char*)(Gx + (size_t)tn * NB * NG);  // 4 ops/wave/step
      char*       ln = (char*)&gxbuf[(t + 1) & 1][0];
#pragma unroll
      for (int i = 0; i < 4; ++i)
        async_copy_b128(ln + i * 16384 + tid * 16,
                        gn + i * 16384 + (size_t)tid * 16);
    }

    // ---- G = h @ Whh^T for columns [nbase, nbase+32), rows 0..31 ----
    const __bf16* a0row = h_bf + (size_t)(lane & 15) * NH + ((lane >> 4) << 3);
    const __bf16* a1row = a0row + 16 * NH;
    v8f acc00 = {}, acc01 = {}, acc10 = {}, acc11 = {};
#pragma unroll
    for (int kk = 0; kk < 8; ++kk) {
      const int k0 = kk * 32;
      v16bf a0 = wmma_a_frag(a0row + k0);
      v16bf a1 = wmma_a_frag(a1row + k0);
      acc00 = wmma_bf16(a0, wB0[kk], acc00);
      acc01 = wmma_bf16(a0, wB1[kk], acc01);
      acc10 = wmma_bf16(a1, wB0[kk], acc10);
      acc11 = wmma_bf16(a1, wB1[kk], acc11);
    }
    const int colb = nbase + (lane & 15);
#pragma unroll
    for (int r = 0; r < 8; ++r) {
      const int row0 = r + ((lane >> 4) << 3);
      gates[row0 * NG + colb]             = acc00[r];
      gates[row0 * NG + colb + 16]        = acc01[r];
      gates[(row0 + 16) * NG + colb]      = acc10[r];
      gates[(row0 + 16) * NG + colb + 16] = acc11[r];
    }

    // async loads complete in order: <=4 outstanding => step-t batch landed
    wait_asynccnt_4();
    __syncthreads();

    // ---- elementwise gate update with length masking (all LDS-resident) ----
    const __bf16* gxl = &gxbuf[t & 1][0];
#pragma unroll
    for (int e = 0; e < 8; ++e) {
      const int idx = tid + e * 1024;        // (b, j) with b = idx/256
      const int b = idx >> 8, j = idx & 255;
      const float*  grow  = gates + b * NG;
      const __bf16* gxrow = gxl + b * NG;
      const float gi = grow[j]          + (float)gxrow[j];
      const float gf = grow[NH + j]     + (float)gxrow[NH + j];
      const float gg = grow[2 * NH + j] + (float)gxrow[2 * NH + j];
      const float go = grow[3 * NH + j] + (float)gxrow[3 * NH + j];
      float cn = sigmoid_(gf) * creg[e] + sigmoid_(gi) * tanh_(gg);
      float hn = sigmoid_(go) * tanh_(cn);
      if (t >= len_s[b]) { cn = creg[e]; hn = hreg[e]; }
      creg[e] = cn; hreg[e] = hn; h_bf[idx] = (__bf16)hn;
      const size_t o = (size_t)t * (2 * NB * NH) + (size_t)(dir * NB + b) * NH + j;
      out_h[o] = hn; out_c[o] = cn;
    }
    __syncthreads();   // h_bf stable before next step's matmul
  }
}

// ---------------------------------------------------------------------------
// Host side
// ---------------------------------------------------------------------------
extern "C" void kernel_launch(void* const* d_in, const int* in_sizes, int n_in,
                              void* d_out, int out_size, void* d_ws, size_t ws_size,
                              hipStream_t stream) {
  (void)in_sizes; (void)n_in; (void)out_size; (void)ws_size;

  const float* x     = (const float*)d_in[0];
  const int*   len   = (const int*)  d_in[1];
  const float* h0    = (const float*)d_in[2];
  const float* c0    = (const float*)d_in[3];
  const float* Wih_f = (const float*)d_in[4];
  const float* Whh_f = (const float*)d_in[5];
  const float* bih_f = (const float*)d_in[6];
  const float* bhh_f = (const float*)d_in[7];
  const float* Wih_b = (const float*)d_in[8];
  const float* Whh_b = (const float*)d_in[9];
  const float* bih_b = (const float*)d_in[10];
  const float* bhh_b = (const float*)d_in[11];
  float* out = (float*)d_out;

  // workspace carve-up (bytes)
  uint8_t* p = (uint8_t*)d_ws;
  __bf16* xb      = (__bf16*)p; p += (size_t)MROWS * NI * 2;        // 32 MB
  __bf16* Wih_fb  = (__bf16*)p; p += (size_t)NG * NI * 2;           // 512 KB
  __bf16* Whh_fb  = (__bf16*)p; p += (size_t)NG * NH * 2;
  __bf16* Wih_bb  = (__bf16*)p; p += (size_t)NG * NI * 2;
  __bf16* Whh_bb  = (__bf16*)p; p += (size_t)NG * NH * 2;
  float*  bsum_f  = (float*)p;  p += (size_t)NG * 4;
  float*  bsum_b  = (float*)p;  p += (size_t)NG * 4;
  __bf16* Gx_f    = (__bf16*)p; p += (size_t)MROWS * NG * 2;        // 128 MB
  __bf16* Gx_b    = (__bf16*)p; p += (size_t)MROWS * NG * 2;        // 128 MB

  // Phase 1: conversions
  lstm_cvt_bf16<<<4096, 256, 0, stream>>>(x, xb, MROWS * NI);
  lstm_cvt_bf16<<<1024, 256, 0, stream>>>(Wih_f, Wih_fb, NG * NI);
  lstm_cvt_bf16<<<1024, 256, 0, stream>>>(Whh_f, Whh_fb, NG * NH);
  lstm_cvt_bf16<<<1024, 256, 0, stream>>>(Wih_b, Wih_bb, NG * NI);
  lstm_cvt_bf16<<<1024, 256, 0, stream>>>(Whh_b, Whh_bb, NG * NH);
  lstm_bias_sum<<<4, 256, 0, stream>>>(bih_f, bhh_f, bsum_f, NG);
  lstm_bias_sum<<<4, 256, 0, stream>>>(bih_b, bhh_b, bsum_b, NG);

  // Phase 2: input projection GEMM (both directions)
  lstm_gemm_x<<<dim3(MROWS / 64, NG / 64, 2), 256, 0, stream>>>(
      xb, Wih_fb, Wih_bb, bsum_f, bsum_b, Gx_f, Gx_b);

  // Phase 3: recurrent scan (one workgroup per direction)
  lstm_scan<<<2, 1024, 0, stream>>>(Gx_f, Gx_b, Whh_fb, Whh_bb, h0, c0, len, out);
}
